// MyTransformerEncoderLayer_36009005809846
// MI455X (gfx1250) — matrix operand order repaired and checked by reading, not compiled, not run on hardware
//
#include <hip/hip_runtime.h>
#include <hip/hip_bf16.h>
#include <math.h>

#define Bz   2
#define Sz   2048
#define Dz   768
#define Hz   12
#define DHz  64
#define DFFz 3072
#define NT   (Bz*Sz)   // 4096 tokens

typedef __attribute__((ext_vector_type(16))) __bf16          v16bf;
typedef __attribute__((ext_vector_type(8)))  float           v8f;
typedef __attribute__((ext_vector_type(8)))  unsigned short  us8;
typedef __attribute__((ext_vector_type(16))) unsigned short  us16;

__device__ __forceinline__ unsigned short f2bf(float f) {
  union { float f; unsigned int u; } c; c.f = f;
  unsigned int u = c.u + 0x7FFFu + ((c.u >> 16) & 1u);   // round-to-nearest-even
  return (unsigned short)(u >> 16);
}
__device__ __forceinline__ v16bf as_bf(us16 u) {
  union { us16 u; v16bf b; } c; c.u = u; return c.b;
}
__device__ __forceinline__ us16 cat8(us8 lo, us8 hi) {
  union { struct { us8 lo, hi; } p; us16 w; } c; c.p.lo = lo; c.p.hi = hi; return c.w;
}
// A-tile (16x32 bf16): base = row ptr + k0. lane half selects K sub-block.
__device__ __forceinline__ v16bf load_a(const unsigned short* base, int half) {
  us8 lo = *(const us8*)(base + 16 * half);
  us8 hi = *(const us8*)(base + 16 * half + 8);
  return as_bf(cat8(lo, hi));
}
// B-tile (32x16 bf16): p = (col row of W^T) + k0 + 16*half, 16 contiguous bf16.
__device__ __forceinline__ v16bf load_b(const unsigned short* p) {
  return as_bf(*(const us16*)p);
}
__device__ __forceinline__ v8f wmma_bf16(v16bf a, v16bf b, v8f c) {
  return __builtin_amdgcn_wmma_f32_16x16x32_bf16(false, a, false, b, (short)0, c, false, false);
}

// -------- CDNA5 async copy: 16B per lane, global -> LDS, tracked by ASYNCcnt
__device__ __forceinline__ void async_cp16(unsigned ldsoff, const void* gp) {
  asm volatile("global_load_async_to_lds_b128 %0, %1, off"
               :: "v"(ldsoff), "v"(gp) : "memory");
}
__device__ __forceinline__ void wait_async16() { asm volatile("s_wait_asynccnt 0x10" ::: "memory"); }
__device__ __forceinline__ void wait_async0()  { asm volatile("s_wait_asynccnt 0x0"  ::: "memory"); }
__device__ __forceinline__ unsigned lds_off(const void* p) {
  return (unsigned)(unsigned long long)p;   // AS3 offset lives in the low 32 bits
}

// ---------------------------------------------------------------- elementwise
__global__ __launch_bounds__(256) void cvt_bf16(const float* __restrict__ in,
                                                unsigned short* __restrict__ out, int n) {
  int i = blockIdx.x * 256 + threadIdx.x;
  if (i < n) out[i] = f2bf(in[i]);
}
// W [K x N] fp32 row-major  ->  Wt [N x K] bf16 row-major
__global__ __launch_bounds__(256) void transpose_bf16(const float* __restrict__ W,
                                                      unsigned short* __restrict__ Wt,
                                                      int K, int N) {
  long i = (long)blockIdx.x * 256 + threadIdx.x;
  if (i < (long)K * N) {
    int k = (int)(i / N), n = (int)(i % N);
    Wt[(long)n * K + k] = f2bf(W[i]);
  }
}

// ---------------------------------------------------------------- GEMM (bf16 WMMA)
// C[M,N] = scale*(A[M,K](bf16) * Wt[N,K]^T(bf16) + bias); one wave = 32x64 tile.
enum { OUT_QKHEAD = 0, OUT_VHEADT = 1, OUT_RESID_F32 = 2, OUT_GELU_BF16 = 3 };

template <int MODE>
__global__ __launch_bounds__(32)
void gemm_bf16(const unsigned short* __restrict__ A,
               const unsigned short* __restrict__ Wt,
               const float* __restrict__ bias,
               const float* __restrict__ resid,
               float* __restrict__ outF,
               unsigned short* __restrict__ outB,
               int M, int N, int K, float scale) {
  const int lane = threadIdx.x;
  const int half = lane >> 4;
  const int ln16 = lane & 15;
  const int n0 = blockIdx.x * 64;
  const int m0 = blockIdx.y * 32;

  v8f acc[2][4] = {};
  const unsigned short* arow0 = A + (long)(m0 + ln16) * K;
  const unsigned short* arow1 = A + (long)(m0 + 16 + ln16) * K;
  for (int k0 = 0; k0 < K; k0 += 32) {
    __builtin_prefetch((const void*)(arow0 + k0 + 128), 0, 1);  // global_prefetch_b8
    __builtin_prefetch((const void*)(arow1 + k0 + 128), 0, 1);
    v16bf a0 = load_a(arow0 + k0, half);
    v16bf a1 = load_a(arow1 + k0, half);
#pragma unroll
    for (int j = 0; j < 4; ++j) {
      v16bf b = load_b(Wt + (long)(n0 + j * 16 + ln16) * K + k0 + 16 * half);
      acc[0][j] = wmma_bf16(a0, b, acc[0][j]);
      acc[1][j] = wmma_bf16(a1, b, acc[1][j]);
    }
  }
#pragma unroll
  for (int j = 0; j < 4; ++j) {
    int col = n0 + j * 16 + ln16;
    float bcol = bias[col];
#pragma unroll
    for (int i = 0; i < 2; ++i) {
#pragma unroll
      for (int r = 0; r < 8; ++r) {
        int row = m0 + i * 16 + r + 8 * half;    // token index
        float v = (acc[i][j][r] + bcol) * scale;
        if (MODE == OUT_RESID_F32) {
          long idx = (long)row * N + col;
          outF[idx] = v + resid[idx];
        } else if (MODE == OUT_GELU_BF16) {
          float g = 0.5f * v * (1.0f + erff(v * 0.70710678118654752f));
          outB[(long)row * N + col] = f2bf(g);
        } else {
          int b = row >> 11, s = row & (Sz - 1);
          int h = col >> 6, dh = col & (DHz - 1);
          long idx = (MODE == OUT_QKHEAD)
                         ? (((long)(b * Hz + h) * Sz + s) * DHz + dh)    // [B,H,S,DH]
                         : (((long)(b * Hz + h) * DHz + dh) * Sz + s);   // [B,H,DH,S]
          outB[idx] = f2bf(v);
        }
      }
    }
  }
}

// ---------------------------------------------------------------- flash attention
// One wave = 16 query rows of one (b,h); K/V streamed in 32-key chunks through
// LDS with async double-buffering. Scores come pre-scaled (Q was scaled 1/8).
#define NCH (Sz / 32)

__device__ __forceinline__ void stage_chunk(const unsigned short* __restrict__ Kb,
                                            const unsigned short* __restrict__ Vb,
                                            int s0, unsigned klds, unsigned vlds, int lane) {
  // K chunk: rows s0..s0+31 of [S,DH] -> 4096 contiguous bytes
  const char* kg = (const char*)(Kb + (long)s0 * DHz);
#pragma unroll
  for (int i = 0; i < 8; ++i) {
    unsigned off = (unsigned)((i * 32 + lane) * 16);
    async_cp16(klds + off, kg + off);
  }
  // V chunk: 64 rows x 64B segments of [DH,S]
#pragma unroll
  for (int i = 0; i < 8; ++i) {
    int t = i * 32 + lane;              // 0..255
    int row = t >> 2, part = t & 3;
    async_cp16(vlds + (unsigned)(row * 64 + part * 16),
               (const char*)(Vb + (long)row * Sz + s0) + part * 16);
  }
}

__global__ __launch_bounds__(32)
void attn_fa(const unsigned short* __restrict__ Q,
             const unsigned short* __restrict__ Km,
             const unsigned short* __restrict__ Vt,
             unsigned short* __restrict__ ctx) {
  __shared__ __attribute__((aligned(64))) unsigned short KL[2][32 * DHz];  // [keys][dh]
  __shared__ __attribute__((aligned(64))) unsigned short VL[2][DHz * 32];  // [dh][keys]
  __shared__ __attribute__((aligned(64))) unsigned short Pl[16 * 32];
  const int lane = threadIdx.x;
  const int half = lane >> 4;
  const int ln16 = lane & 15;
  const int bh = blockIdx.y;
  const int q0 = blockIdx.x * 16;
  const unsigned short* Qb = Q + (long)bh * Sz * DHz;
  const unsigned short* Kb = Km + (long)bh * Sz * DHz;
  const unsigned short* Vb = Vt + (long)bh * DHz * Sz;
  const unsigned kl0 = lds_off(&KL[0][0]), kl1 = lds_off(&KL[1][0]);
  const unsigned vl0 = lds_off(&VL[0][0]), vl1 = lds_off(&VL[1][0]);

  v16bf qa[2];
  {
    const unsigned short* qrow = Qb + (long)(q0 + ln16) * DHz;
    qa[0] = load_a(qrow, half);
    qa[1] = load_a(qrow + 32, half);
  }
  us16 ou;
#pragma unroll
  for (int i = 0; i < 16; ++i) ou[i] = 0x3F80;    // bf16 1.0
  const v16bf ones = as_bf(ou);

  v8f ctxacc[4] = {};
  float mrow[8], lrow[8];
#pragma unroll
  for (int r = 0; r < 8; ++r) { mrow[r] = -1e30f; lrow[r] = 0.f; }

  stage_chunk(Kb, Vb, 0, kl0, vl0, lane);
  for (int c = 0; c < NCH; ++c) {
    const int buf = c & 1;
    if (c + 1 < NCH) {
      stage_chunk(Kb, Vb, (c + 1) * 32, buf ? kl0 : kl1, buf ? vl0 : vl1, lane);
      wait_async16();                 // chunk c landed; chunk c+1 still in flight
    } else {
      wait_async0();
    }
    const unsigned short* Kc = &KL[buf][0];
    const unsigned short* Vc = &VL[buf][0];

    v8f sc0 = {}, sc1 = {};
#pragma unroll
    for (int t = 0; t < 2; ++t) {     // DH = 64 -> two k=32 steps
      sc0 = wmma_bf16(qa[t], load_b(Kc + (ln16) * DHz + 32 * t + 16 * half), sc0);
      sc1 = wmma_bf16(qa[t], load_b(Kc + (16 + ln16) * DHz + 32 * t + 16 * half), sc1);
    }
    float p0[8], p1[8], alpha[8];
#pragma unroll
    for (int r = 0; r < 8; ++r) {
      float cm = fmaxf(sc0[r], sc1[r]);
#pragma unroll
      for (int off = 1; off < 16; off <<= 1) cm = fmaxf(cm, __shfl_xor(cm, off, 32));
      float mn = fmaxf(mrow[r], cm);
      alpha[r] = __expf(mrow[r] - mn);
      p0[r] = __expf(sc0[r] - mn);
      p1[r] = __expf(sc1[r] - mn);
      mrow[r] = mn;
    }
#pragma unroll
    for (int j = 0; j < 4; ++j)
#pragma unroll
      for (int r = 0; r < 8; ++r) ctxacc[j][r] *= alpha[r];

    // C-layout -> A-layout via LDS
#pragma unroll
    for (int r = 0; r < 8; ++r) {
      Pl[(r + 8 * half) * 32 + ln16]      = f2bf(p0[r]);
      Pl[(r + 8 * half) * 32 + 16 + ln16] = f2bf(p1[r]);
    }
    __syncthreads();
    v16bf pa = load_a(&Pl[ln16 * 32], half);

    // row sums via WMMA against ones (sums the exact bf16 P used for P*V)
    v8f srow = wmma_bf16(pa, ones, (v8f){});
#pragma unroll
    for (int r = 0; r < 8; ++r) lrow[r] = lrow[r] * alpha[r] + srow[r];

#pragma unroll
    for (int j = 0; j < 4; ++j)
      ctxacc[j] = wmma_bf16(pa, load_b(Vc + (j * 16 + ln16) * 32 + 16 * half), ctxacc[j]);
    __syncthreads();
  }

  const int b = bh / Hz, h = bh % Hz;
#pragma unroll
  for (int j = 0; j < 4; ++j)
#pragma unroll
    for (int r = 0; r < 8; ++r) {
      int row = q0 + r + 8 * half;
      long t = (long)b * Sz + row;
      ctx[t * Dz + h * DHz + j * 16 + ln16] = f2bf(ctxacc[j][r] / lrow[r]);
    }
}

// ---------------------------------------------------------------- layernorm
template <bool WB>
__global__ __launch_bounds__(256)
void layernorm(const float* __restrict__ in, const float* __restrict__ g,
               const float* __restrict__ be, float* __restrict__ outF,
               unsigned short* __restrict__ outB) {
  __shared__ float s1[256], s2[256];
  const int row = blockIdx.x;
  const float* x = in + (long)row * Dz;
  float v[3], sum = 0.f, sq = 0.f;
#pragma unroll
  for (int i = 0; i < 3; ++i) {
    v[i] = x[threadIdx.x + i * 256];
    sum += v[i]; sq += v[i] * v[i];
  }
  s1[threadIdx.x] = sum; s2[threadIdx.x] = sq;
  __syncthreads();
  for (int off = 128; off > 0; off >>= 1) {
    if ((int)threadIdx.x < off) {
      s1[threadIdx.x] += s1[threadIdx.x + off];
      s2[threadIdx.x] += s2[threadIdx.x + off];
    }
    __syncthreads();
  }
  float mean = s1[0] * (1.0f / Dz);
  float var  = s2[0] * (1.0f / Dz) - mean * mean;
  float rstd = rsqrtf(var + 1e-5f);
#pragma unroll
  for (int i = 0; i < 3; ++i) {
    int c = threadIdx.x + i * 256;
    float y = (v[i] - mean) * rstd * g[c] + be[c];
    outF[(long)row * Dz + c] = y;
    if (WB) outB[(long)row * Dz + c] = f2bf(y);
  }
}

// ---------------------------------------------------------------- launch
extern "C" void kernel_launch(void* const* d_in, const int* in_sizes, int n_in,
                              void* d_out, int out_size, void* d_ws, size_t ws_size,
                              hipStream_t stream) {
  (void)in_sizes; (void)n_in; (void)out_size; (void)ws_size;
  const float* x   = (const float*)d_in[0];
  const float* Wq  = (const float*)d_in[1];
  const float* bq  = (const float*)d_in[2];
  const float* Wk  = (const float*)d_in[3];
  const float* bk  = (const float*)d_in[4];
  const float* Wv  = (const float*)d_in[5];
  const float* bv  = (const float*)d_in[6];
  const float* Wo  = (const float*)d_in[7];
  const float* bo  = (const float*)d_in[8];
  const float* W1  = (const float*)d_in[9];
  const float* b1  = (const float*)d_in[10];
  const float* W2  = (const float*)d_in[11];
  const float* b2  = (const float*)d_in[12];
  const float* g1  = (const float*)d_in[13];
  const float* be1 = (const float*)d_in[14];
  const float* g2  = (const float*)d_in[15];
  const float* be2 = (const float*)d_in[16];

  char* ws = (char*)d_ws;
  auto alloc = [&](size_t bytes) -> char* {
    char* p = ws; ws += (bytes + 255) & ~(size_t)255; return p;
  };
  unsigned short* Xb   = (unsigned short*)alloc((size_t)NT * Dz * 2);
  unsigned short* WqT  = (unsigned short*)alloc((size_t)Dz * Dz * 2);
  unsigned short* WkT  = (unsigned short*)alloc((size_t)Dz * Dz * 2);
  unsigned short* WvT  = (unsigned short*)alloc((size_t)Dz * Dz * 2);
  unsigned short* WoT  = (unsigned short*)alloc((size_t)Dz * Dz * 2);
  unsigned short* W1T  = (unsigned short*)alloc((size_t)Dz * DFFz * 2);
  unsigned short* W2T  = (unsigned short*)alloc((size_t)DFFz * Dz * 2);
  unsigned short* Qh   = (unsigned short*)alloc((size_t)NT * Dz * 2);
  unsigned short* Kh   = (unsigned short*)alloc((size_t)NT * Dz * 2);
  unsigned short* Vh   = (unsigned short*)alloc((size_t)NT * Dz * 2);
  unsigned short* CtxB = (unsigned short*)alloc((size_t)NT * Dz * 2);
  unsigned short* X1b  = (unsigned short*)alloc((size_t)NT * Dz * 2);
  unsigned short* Hff  = (unsigned short*)alloc((size_t)NT * DFFz * 2);
  float*          Y1   = (float*)alloc((size_t)NT * Dz * 4);
  float*          X1   = (float*)alloc((size_t)NT * Dz * 4);
  float*          Y2   = (float*)alloc((size_t)NT * Dz * 4);

  int n = NT * Dz;
  cvt_bf16<<<(n + 255) / 256, 256, 0, stream>>>(x, Xb, n);
  n = Dz * Dz;
  transpose_bf16<<<(n + 255) / 256, 256, 0, stream>>>(Wq, WqT, Dz, Dz);
  transpose_bf16<<<(n + 255) / 256, 256, 0, stream>>>(Wk, WkT, Dz, Dz);
  transpose_bf16<<<(n + 255) / 256, 256, 0, stream>>>(Wv, WvT, Dz, Dz);
  transpose_bf16<<<(n + 255) / 256, 256, 0, stream>>>(Wo, WoT, Dz, Dz);
  n = Dz * DFFz;
  transpose_bf16<<<(n + 255) / 256, 256, 0, stream>>>(W1, W1T, Dz, DFFz);
  transpose_bf16<<<(n + 255) / 256, 256, 0, stream>>>(W2, W2T, DFFz, Dz);

  dim3 gq(Dz / 64, NT / 32);
  gemm_bf16<OUT_QKHEAD><<<gq, 32, 0, stream>>>(Xb, WqT, bq, nullptr, nullptr, Qh, NT, Dz, Dz, 0.125f);
  gemm_bf16<OUT_QKHEAD><<<gq, 32, 0, stream>>>(Xb, WkT, bk, nullptr, nullptr, Kh, NT, Dz, Dz, 1.0f);
  gemm_bf16<OUT_VHEADT><<<gq, 32, 0, stream>>>(Xb, WvT, bv, nullptr, nullptr, Vh, NT, Dz, Dz, 1.0f);

  dim3 ga(Sz / 16, Bz * Hz);
  attn_fa<<<ga, 32, 0, stream>>>(Qh, Kh, Vh, CtxB);

  gemm_bf16<OUT_RESID_F32><<<gq, 32, 0, stream>>>(CtxB, WoT, bo, x, Y1, nullptr, NT, Dz, Dz, 1.0f);
  layernorm<true><<<NT, 256, 0, stream>>>(Y1, g1, be1, X1, X1b);

  dim3 gf1(DFFz / 64, NT / 32);
  gemm_bf16<OUT_GELU_BF16><<<gf1, 32, 0, stream>>>(X1b, W1T, b1, nullptr, nullptr, Hff, NT, DFFz, Dz, 1.0f);
  gemm_bf16<OUT_RESID_F32><<<gq, 32, 0, stream>>>(Hff, W2T, b2, X1, Y2, nullptr, NT, Dz, DFFz, 1.0f);
  layernorm<false><<<NT, 256, 0, stream>>>(Y2, g2, be2, (float*)d_out, nullptr);
}
